// MixtralMoE_70016556860060
// MI455X (gfx1250) — matrix operand order repaired and checked by reading, not compiled, not run on hardware
//
#include <hip/hip_runtime.h>
#include <hip/hip_bf16.h>

#define NEXP 8
#define HID  2048
#define INTR 7168
#define NTOK 8192

typedef __attribute__((ext_vector_type(16))) __bf16 v16bf;
typedef __attribute__((ext_vector_type(2)))  __bf16 v2bf;
typedef __attribute__((ext_vector_type(8)))  float  v8f;

// ---- f32 -> bf16 conversion helpers ----
static __device__ __forceinline__ unsigned short f2bf(float f) {
  union { float f; unsigned u; } c; c.f = f;
  unsigned u = c.u;
  return (unsigned short)((u + 0x7FFFu + ((u >> 16) & 1u)) >> 16);
}

// pack two f32 -> bf16x2 in one dword
#if __has_builtin(__builtin_amdgcn_cvt_pk_bf16_f32)
static __device__ __forceinline__ unsigned pk2(float a, float b) {
  union { v2bf v; unsigned u; } c;
  c.v = __builtin_amdgcn_cvt_pk_bf16_f32(a, b);
  return c.u;
}
#elif __has_builtin(__builtin_amdgcn_perm)
static __device__ __forceinline__ unsigned pk2(float a, float b) {
  union { float f; unsigned u; } ca, cb;
  ca.f = a; cb.f = b;
  // one v_perm_b32: {b[31:16], a[31:16]}  (truncation rounding)
  return __builtin_amdgcn_perm(cb.u, ca.u, 0x07060302u);
}
#else
static __device__ __forceinline__ unsigned pk2(float a, float b) {
  return (unsigned)f2bf(a) | ((unsigned)f2bf(b) << 16);
}
#endif

static __device__ __forceinline__ uint4 pack8(float4 a, float4 b) {
  uint4 r;
  r.x = pk2(a.x, a.y);
  r.y = pk2(a.z, a.w);
  r.z = pk2(b.x, b.y);
  r.w = pk2(b.z, b.w);
  return r;
}

static __device__ __forceinline__ v8f vzero() {
  v8f z;
#pragma unroll
  for (int i = 0; i < 8; ++i) z[i] = 0.f;
  return z;
}

// ---- CDNA5 async global->LDS copy (probe-verified signature) ----
#if __has_builtin(__builtin_amdgcn_global_load_async_to_lds_b128)
#define HAVE_ASYNC_LDS 1
typedef int vint4 __attribute__((vector_size(16)));
typedef __attribute__((address_space(1))) vint4* gv4_ptr;  // global
typedef __attribute__((address_space(3))) vint4* lv4_ptr;  // LDS
static __device__ __forceinline__ void async_cp16(const void* g, void* l) {
  __builtin_amdgcn_global_load_async_to_lds_b128((gv4_ptr)g, (lv4_ptr)l, 0, 0);
}
static __device__ __forceinline__ void async_wait() {
#if __has_builtin(__builtin_amdgcn_s_wait_asynccnt)
  __builtin_amdgcn_s_wait_asynccnt(0);
#else
  asm volatile("s_wait_asynccnt 0" ::: "memory");
#endif
}
#else
#define HAVE_ASYNC_LDS 0
#endif

// ---------------- zero out + counters ----------------
__global__ void moe_zero(float* __restrict__ out, int* __restrict__ counts, long n) {
  long gid = (long)blockIdx.x * blockDim.x + threadIdx.x;
  long stride = (long)gridDim.x * blockDim.x;
  for (long i = gid; i < n; i += stride) out[i] = 0.f;
  if (gid < NEXP) counts[gid] = 0;
}

// ---------------- f32 -> bf16 activation convert ----------------
__global__ void moe_cvt(const float* __restrict__ x, unsigned short* __restrict__ xb, long n4) {
  long gid = (long)blockIdx.x * blockDim.x + threadIdx.x;
  long stride = (long)gridDim.x * blockDim.x;
  for (long i = gid; i < n4; i += stride) {
    float4 f = ((const float4*)x)[i];
    uint2 r;
    r.x = pk2(f.x, f.y);
    r.y = pk2(f.z, f.w);
    ((uint2*)xb)[i] = r;
  }
}

// ---------------- router: logits -> softmax -> top2 -> expert lists ----------------
__global__ __launch_bounds__(64) void moe_router(const float* __restrict__ x,
                                                 const float* __restrict__ gw,
                                                 int* __restrict__ rows,
                                                 float* __restrict__ wts,
                                                 int* __restrict__ counts) {
  const int t = blockIdx.x;
  const int tid = threadIdx.x;
  const float* xp = x + (size_t)t * HID;
  float l[NEXP];
#pragma unroll
  for (int e = 0; e < NEXP; ++e) l[e] = 0.f;
  for (int h = tid; h < HID; h += 64) {
    float xv = xp[h];
#pragma unroll
    for (int e = 0; e < NEXP; ++e) l[e] += xv * gw[e * HID + h];
  }
  __shared__ float part[64][NEXP];
#pragma unroll
  for (int e = 0; e < NEXP; ++e) part[tid][e] = l[e];
  __syncthreads();
  if (tid == 0) {
    float lg[NEXP];
    for (int e = 0; e < NEXP; ++e) {
      float s = 0.f;
      for (int j = 0; j < 64; ++j) s += part[j][e];
      lg[e] = s;
    }
    float mx = lg[0];
    for (int e = 1; e < NEXP; ++e) mx = fmaxf(mx, lg[e]);
    float p[NEXP];
    for (int e = 0; e < NEXP; ++e) p[e] = __expf(lg[e] - mx);
    int i0 = 0;
    for (int e = 1; e < NEXP; ++e) if (p[e] > p[i0]) i0 = e;  // first-index wins ties
    int i1 = (i0 == 0) ? 1 : 0;
    for (int e = 0; e < NEXP; ++e) if (e != i0 && e != i1 && p[e] > p[i1]) i1 = e;
    float denom = p[i0] + p[i1];
    int pos0 = atomicAdd(&counts[i0], 1);
    rows[i0 * NTOK + pos0] = t;
    wts [i0 * NTOK + pos0] = p[i0] / denom;
    int pos1 = atomicAdd(&counts[i1], 1);
    rows[i1 * NTOK + pos1] = t;
    wts [i1 * NTOK + pos1] = p[i1] / denom;
  }
}

// ---------------- GEMM1: gated = silu(x@w1^T) * (x@w3^T) for one expert ----------------
// block tile 128M x 64N, K step 32, bf16 WMMA, gathered A rows.
// A staged via async global->LDS DMA; B (f32 weights) software-pipelined through
// registers and converted to bf16 with v_perm on the way into LDS.
__global__ __launch_bounds__(256) void moe_gemm1(const unsigned short* __restrict__ xb,
                                                 const float* __restrict__ w1e,
                                                 const float* __restrict__ w3e,
                                                 const int* __restrict__ rows,
                                                 const int* __restrict__ counts,
                                                 unsigned short* __restrict__ gated,
                                                 int e) {
  const int cnt = counts[e];
  const int m_blk = blockIdx.y * 128;
  if (m_blk >= cnt) return;
  const int n0 = blockIdx.x * 64;
  const int tid = threadIdx.x;

  __shared__ __align__(32) unsigned short As [128][32];
  __shared__ __align__(32) unsigned short B1s[64][32];
  __shared__ __align__(32) unsigned short B3s[64][32];

  // A-staging: thread -> (row, 16-element half)
  const int rA = tid >> 1;
  const int hA = tid & 1;
  const int rm_ld = m_blk + rA;
  const int ridx = rm_ld < cnt ? rm_ld : cnt - 1;
  const int tokenA = rows[e * NTOK + ridx] & (NTOK - 1);
  const unsigned short* aSrc = xb + (size_t)tokenA * HID + hA * 16;

  // B-staging: thread -> (row n, 8-float quarter)
  const int nB = tid >> 2;
  const int qB = tid & 3;
  const float* b1Src = w1e + (size_t)(n0 + nB) * HID + qB * 8;
  const float* b3Src = w3e + (size_t)(n0 + nB) * HID + qB * 8;

  const int lane  = tid & 31;
  const int wid   = tid >> 5;
  const int waveM = wid & 3;   // 4 groups of 32 rows
  const int waveN = wid >> 2;  // 2 groups of 32 cols
  const int lh = lane >> 4;
  const int lc = lane & 15;

  v8f acc1[2][2], acc3[2][2];
#pragma unroll
  for (int mi = 0; mi < 2; ++mi)
#pragma unroll
    for (int ni = 0; ni < 2; ++ni) { acc1[mi][ni] = vzero(); acc3[mi][ni] = vzero(); }

  // pipeline prologue: first B tile into registers
  float4 f10 = *(const float4*)(b1Src);
  float4 f11 = *(const float4*)(b1Src + 4);
  float4 f30 = *(const float4*)(b3Src);
  float4 f31 = *(const float4*)(b3Src + 4);

  for (int k0 = 0; k0 < HID; k0 += 32) {
#if HAVE_ASYNC_LDS
    async_cp16(aSrc + k0,     &As[rA][hA * 16]);
    async_cp16(aSrc + k0 + 8, &As[rA][hA * 16 + 8]);
#else
    *(uint4*)&As[rA][hA * 16]     = *(const uint4*)(aSrc + k0);
    *(uint4*)&As[rA][hA * 16 + 8] = *(const uint4*)(aSrc + k0 + 8);
#endif
    // convert+store current B tile (regs), then issue next tile's loads
    *(uint4*)&B1s[nB][qB * 8] = pack8(f10, f11);
    *(uint4*)&B3s[nB][qB * 8] = pack8(f30, f31);
    if (k0 + 32 < HID) {
      f10 = *(const float4*)(b1Src + k0 + 32);
      f11 = *(const float4*)(b1Src + k0 + 36);
      f30 = *(const float4*)(b3Src + k0 + 32);
      f31 = *(const float4*)(b3Src + k0 + 36);
    }
#if HAVE_ASYNC_LDS
    async_wait();
#endif
    __syncthreads();

    union AF { uint4 p[2]; v16bf v; } a[2];
#pragma unroll
    for (int mi = 0; mi < 2; ++mi) {
      const unsigned short* ap = &As[waveM * 32 + mi * 16 + lc][lh * 16];
      a[mi].p[0] = *(const uint4*)ap;
      a[mi].p[1] = *(const uint4*)(ap + 8);
    }
#pragma unroll
    for (int ni = 0; ni < 2; ++ni) {
      v16bf b1 = *(const v16bf*)&B1s[waveN * 32 + ni * 16 + lc][lh * 16];
      v16bf b3 = *(const v16bf*)&B3s[waveN * 32 + ni * 16 + lc][lh * 16];
#pragma unroll
      for (int mi = 0; mi < 2; ++mi) {
        acc1[mi][ni] = __builtin_amdgcn_wmma_f32_16x16x32_bf16(
            false, a[mi].v, false, b1, (short)0, acc1[mi][ni], false, false);
        acc3[mi][ni] = __builtin_amdgcn_wmma_f32_16x16x32_bf16(
            false, a[mi].v, false, b3, (short)0, acc3[mi][ni], false, false);
      }
    }
    __syncthreads();
  }

  // epilogue: silu(acc1)*acc3 -> bf16 gated buffer (expert-local rows)
#pragma unroll
  for (int mi = 0; mi < 2; ++mi)
#pragma unroll
    for (int ni = 0; ni < 2; ++ni)
#pragma unroll
      for (int i = 0; i < 8; ++i) {
        int rm = m_blk + waveM * 32 + mi * 16 + i + 8 * lh;
        if (rm < cnt) {
          int n = n0 + waveN * 32 + ni * 16 + lc;
          float g1 = acc1[mi][ni][i];
          float g3 = acc3[mi][ni][i];
          float s = g1 / (1.f + __expf(-g1));
          gated[(size_t)rm * INTR + n] = f2bf(s * g3);
        }
      }
}

// ---------------- GEMM2: out[token] += weight * (gated @ w2^T) ----------------
__global__ __launch_bounds__(256) void moe_gemm2(const unsigned short* __restrict__ gated,
                                                 const float* __restrict__ w2e,
                                                 const int* __restrict__ rows,
                                                 const float* __restrict__ wts,
                                                 const int* __restrict__ counts,
                                                 float* __restrict__ out,
                                                 int e) {
  const int cnt = counts[e];
  const int m_blk = blockIdx.y * 128;
  if (m_blk >= cnt) return;
  const int n0 = blockIdx.x * 64;
  const int tid = threadIdx.x;

  __shared__ __align__(32) unsigned short As[128][32];
  __shared__ __align__(32) unsigned short Bs[64][32];

  const int rA = tid >> 1;
  const int hA = tid & 1;
  const unsigned short* aSrc = gated + (size_t)(m_blk + rA) * INTR + hA * 16;

  const int nB = tid >> 2;
  const int qB = tid & 3;
  const float* bSrc = w2e + (size_t)(n0 + nB) * INTR + qB * 8;

  const int lane  = tid & 31;
  const int wid   = tid >> 5;
  const int waveM = wid & 3;
  const int waveN = wid >> 2;
  const int lh = lane >> 4;
  const int lc = lane & 15;

  v8f acc[2][2];
#pragma unroll
  for (int mi = 0; mi < 2; ++mi)
#pragma unroll
    for (int ni = 0; ni < 2; ++ni) acc[mi][ni] = vzero();

  float4 f0 = *(const float4*)(bSrc);
  float4 f1 = *(const float4*)(bSrc + 4);

  for (int k0 = 0; k0 < INTR; k0 += 32) {
#if HAVE_ASYNC_LDS
    async_cp16(aSrc + k0,     &As[rA][hA * 16]);
    async_cp16(aSrc + k0 + 8, &As[rA][hA * 16 + 8]);
#else
    *(uint4*)&As[rA][hA * 16]     = *(const uint4*)(aSrc + k0);
    *(uint4*)&As[rA][hA * 16 + 8] = *(const uint4*)(aSrc + k0 + 8);
#endif
    *(uint4*)&Bs[nB][qB * 8] = pack8(f0, f1);
    if (k0 + 32 < INTR) {
      f0 = *(const float4*)(bSrc + k0 + 32);
      f1 = *(const float4*)(bSrc + k0 + 36);
    }
#if HAVE_ASYNC_LDS
    async_wait();
#endif
    __syncthreads();

    union AF { uint4 p[2]; v16bf v; } a[2];
#pragma unroll
    for (int mi = 0; mi < 2; ++mi) {
      const unsigned short* ap = &As[waveM * 32 + mi * 16 + lc][lh * 16];
      a[mi].p[0] = *(const uint4*)ap;
      a[mi].p[1] = *(const uint4*)(ap + 8);
    }
#pragma unroll
    for (int ni = 0; ni < 2; ++ni) {
      v16bf b = *(const v16bf*)&Bs[waveN * 32 + ni * 16 + lc][lh * 16];
#pragma unroll
      for (int mi = 0; mi < 2; ++mi) {
        acc[mi][ni] = __builtin_amdgcn_wmma_f32_16x16x32_bf16(
            false, a[mi].v, false, b, (short)0, acc[mi][ni], false, false);
      }
    }
    __syncthreads();
  }

  // epilogue: scale by routing weight, accumulate into out (rows unique within expert;
  // experts run sequentially on the stream, so plain RMW is safe)
#pragma unroll
  for (int mi = 0; mi < 2; ++mi)
#pragma unroll
    for (int i = 0; i < 8; ++i) {
      int rm = m_blk + waveM * 32 + mi * 16 + i + 8 * lh;
      if (rm < cnt) {
        int tok  = rows[e * NTOK + rm] & (NTOK - 1);
        float wv = wts[e * NTOK + rm];
#pragma unroll
        for (int ni = 0; ni < 2; ++ni) {
          int n = n0 + waveN * 32 + ni * 16 + lc;
          float* o = out + (size_t)tok * HID + n;
          *o += wv * acc[mi][ni][i];
        }
      }
    }
}

extern "C" void kernel_launch(void* const* d_in, const int* in_sizes, int n_in,
                              void* d_out, int out_size, void* d_ws, size_t ws_size,
                              hipStream_t stream) {
  (void)in_sizes; (void)n_in; (void)out_size; (void)ws_size;
  const float* x  = (const float*)d_in[0];
  const float* gw = (const float*)d_in[1];
  const float* w1 = (const float*)d_in[2];
  const float* w3 = (const float*)d_in[3];
  const float* w2 = (const float*)d_in[4];
  float* out = (float*)d_out;

  // workspace layout
  char* p = (char*)d_ws;
  unsigned short* xb = (unsigned short*)p; p += (size_t)NTOK * HID * 2;   // 32 MB bf16 activations
  int*   rows   = (int*)p;                 p += (size_t)NEXP * NTOK * 4;  // per-expert token lists
  float* wts    = (float*)p;               p += (size_t)NEXP * NTOK * 4;  // per-expert routing weights
  int*   counts = (int*)p;                 p += 256;                      // per-expert counts
  unsigned short* gated = (unsigned short*)p;                             // 117 MB bf16, reused per expert

  moe_zero<<<2048, 256, 0, stream>>>(out, counts, (long)NTOK * HID);
  moe_cvt <<<2048, 256, 0, stream>>>(x, xb, (long)NTOK * HID / 4);
  moe_router<<<NTOK, 64, 0, stream>>>(x, gw, rows, wts, counts);

  for (int e = 0; e < NEXP; ++e) {
    moe_gemm1<<<dim3(INTR / 64, NTOK / 128), 256, 0, stream>>>(
        xb, w1 + (size_t)e * INTR * HID, w3 + (size_t)e * INTR * HID,
        rows, counts, gated, e);
    moe_gemm2<<<dim3(HID / 64, NTOK / 128), 256, 0, stream>>>(
        gated, w2 + (size_t)e * HID * INTR, rows, wts, counts, out, e);
  }
}
